// Stem_32169305047349
// MI455X (gfx1250) — matrix operand order repaired and checked by reading, not compiled, not run on hardware
//
#include <hip/hip_runtime.h>

typedef _Float16 v16h __attribute__((ext_vector_type(16)));
typedef _Float16 v8h  __attribute__((ext_vector_type(8)));
typedef float    v8f  __attribute__((ext_vector_type(8)));

// problem constants (compile-time so no runtime div/mod sequences)
constexpr int CB  = 8;        // batch
constexpr int CN1 = 50176;
constexpr int CN2 = 12544;    // % 16 == 0
constexpr int CN3 = 3136;
constexpr int CM  = CB * CN2; // 100352 GEMM rows

// ---------------------------------------------------------------------------
// 0) zero a float buffer (stats must be zeroed every call; ws is poisoned)
// ---------------------------------------------------------------------------
__global__ void zero_f32(float* p, int n) {
    int i = blockIdx.x * blockDim.x + threadIdx.x;
    if (i < n) p[i] = 0.f;
}

// ---------------------------------------------------------------------------
// 1) offset-MLP: a[n,k,t] = relu(off @ Wh + bh) @ Wo + bo
//    one thread per (n,k); a stored as [n*81 + k*9 + t]
// ---------------------------------------------------------------------------
__global__ void mlp_a(const float* __restrict__ off,
                      const float* __restrict__ Wh, const float* __restrict__ bh,
                      const float* __restrict__ Wo, const float* __restrict__ bo,
                      float* __restrict__ a, int total) {
    int i = blockIdx.x * blockDim.x + threadIdx.x;
    if (i >= total) return;
    float o0 = off[2 * i], o1 = off[2 * i + 1];
    float out[9];
#pragma unroll
    for (int t = 0; t < 9; ++t) out[t] = bo[t];
#pragma unroll
    for (int j = 0; j < 32; ++j) {
        float h = o0 * Wh[j] + o1 * Wh[32 + j] + bh[j];
        h = h > 0.f ? h : 0.f;
#pragma unroll
        for (int t = 0; t < 9; ++t) out[t] += h * Wo[j * 9 + t];
    }
#pragma unroll
    for (int t = 0; t < 9; ++t) a[i * 9 + t] = out[t];
}

// ---------------------------------------------------------------------------
// 2) convert conv weight (O, Kin) fp32 -> (O, Kld) fp16, zero-padded K
// ---------------------------------------------------------------------------
__global__ void convert_w(const float* __restrict__ W, _Float16* __restrict__ Wf,
                          int Kin, int Kld, int total) {
    int i = blockIdx.x * blockDim.x + threadIdx.x;
    if (i >= total) return;
    int o = i / Kld, k = i % Kld;
    Wf[i] = (k < Kin) ? (_Float16)W[o * Kin + k] : (_Float16)0.f;
}

// ---------------------------------------------------------------------------
// 3a) layer-1 gather: g[(b*N2+n)*32 + c*9+t] = sum_k x[b,c,nbr1[n,k]] * a[n,k,t]
//     block = 32 threads per point n (B*C = 24 active), pads cols 27..31
// ---------------------------------------------------------------------------
__global__ void gather_g1(const float* __restrict__ x, const int* __restrict__ nbr,
                          const float* __restrict__ a, _Float16* __restrict__ g) {
    __shared__ float sA[81];
    __shared__ int   sI[9];
    int n = blockIdx.x, tid = threadIdx.x;
    for (int i = tid; i < 81; i += 32) sA[i] = a[n * 81 + i];
    if (tid < 9) sI[tid] = nbr[n * 9 + tid];
    __syncthreads();
    // zero padding columns 27..31 for all 8 rows of this n
    for (int i = tid; i < 40; i += 32) {
        int b = i / 5;
        g[((size_t)(b * CN2 + n)) * 32 + 27 + (i % 5)] = (_Float16)0.f;
    }
    if (tid < 24) {
        int b = tid / 3, c = tid % 3;
        const float* xp = x + (size_t)(b * 3 + c) * CN1;
        float f[9];
#pragma unroll
        for (int k = 0; k < 9; ++k) f[k] = xp[sI[k]];
        _Float16* gp = g + ((size_t)(b * CN2 + n)) * 32 + c * 9;
#pragma unroll
        for (int t = 0; t < 9; ++t) {
            float s = 0.f;
#pragma unroll
            for (int k = 0; k < 9; ++k) s += f[k] * sA[k * 9 + t];
            gp[t] = (_Float16)s;
        }
    }
}

// ---------------------------------------------------------------------------
// 3b) layer-2/3 gather with fused BN+ReLU on the input:
//     feats = relu(y*scale[c]+shift[c]);  g row stride 288, C = 32
//     block = 256 threads = (b in 0..7) x (c in 0..31), one point n per block
// ---------------------------------------------------------------------------
__global__ void gather_g32(const float* __restrict__ y, const int* __restrict__ nbr,
                           const float* __restrict__ a,
                           const float* __restrict__ scale, const float* __restrict__ shift,
                           _Float16* __restrict__ g) {
    __shared__ float sA[81];
    __shared__ int   sI[9];
    int n = blockIdx.x, tid = threadIdx.x;
    if (tid < 81) sA[tid] = a[n * 81 + tid];
    if (tid >= 96 && tid < 105) sI[tid - 96] = nbr[n * 9 + (tid - 96)];
    __syncthreads();
    int b = tid >> 5, c = tid & 31;
    float sc = scale[c], sh = shift[c];
    const float* yp = y + (size_t)(b * 32 + c) * CN2;
    float f[9];
#pragma unroll
    for (int k = 0; k < 9; ++k) {
        float v = yp[sI[k]] * sc + sh;
        f[k] = v > 0.f ? v : 0.f;
    }
    _Float16* gp = g + ((size_t)(b * CN2 + n)) * 288 + c * 9;
#pragma unroll
    for (int t = 0; t < 9; ++t) {
        float s = 0.f;
#pragma unroll
        for (int k = 0; k < 9; ++k) s += f[k] * sA[k * 9 + t];
        gp[t] = (_Float16)s;
    }
}

// ---------------------------------------------------------------------------
// 4) WMMA GEMM:  y[b,o,n] = sum_k g[row,k] * Wf[o,k] + bias[o]
//    row = b*N2+n, M = B*N2, K = KCH*32, N = O = NT*16.
//    One wave computes a 16-row stripe across ALL NT column tiles: the A
//    fragment (streamed g operand) is loaded once per K-chunk and reused for
//    NT WMMAs. Epilogue fuses bias + per-channel sum/sumsq for BN.
// ---------------------------------------------------------------------------
template <int KCH, int KLD, int NT>
__global__ void gemm_wmma(const _Float16* __restrict__ g, const _Float16* __restrict__ wf,
                          const float* __restrict__ bias, float* __restrict__ y,
                          float* __restrict__ sum, float* __restrict__ sumsq) {
    constexpr int O = NT * 16;
    int wave = threadIdx.x >> 5;
    int lane = threadIdx.x & 31;
    int mtile = blockIdx.x * (blockDim.x >> 5) + wave;
    int rowBase = mtile * 16;
    int half = lane >> 4;   // 0: lanes 0-15, 1: lanes 16-31
    int l    = lane & 15;

    v8f acc[NT];
#pragma unroll
    for (int nt = 0; nt < NT; ++nt) acc[nt] = v8f{};

    // A fragment per lane (16-bit A 16x32 layout): row = rowBase+l,
    //   K = half*8 + {0..7}  and  half*8 + 16 + {0..7}
    const _Float16* ga = g  + (size_t)(rowBase + l) * KLD + half * 8;
    // B fragment (16-bit B 32x16 layout): col = nt*16+l,
    //   K = half*8 + {0..7}  and  half*8 + 16 + {0..7}
    const _Float16* wb = wf + (size_t)l * KLD + half * 8;

#pragma unroll
    for (int kc = 0; kc < KCH; ++kc) {
        v8h a0 = *(const v8h*)(ga + kc * 32);
        v8h a1 = *(const v8h*)(ga + kc * 32 + 16);
        v16h afrag = __builtin_shufflevector(a0, a1, 0, 1, 2, 3, 4, 5, 6, 7,
                                             8, 9, 10, 11, 12, 13, 14, 15);
#pragma unroll
        for (int nt = 0; nt < NT; ++nt) {
            const _Float16* wbt = wb + (size_t)nt * 16 * KLD + kc * 32;
            v8h b0 = *(const v8h*)(wbt);
            v8h b1 = *(const v8h*)(wbt + 16);
            v16h bfrag = __builtin_shufflevector(b0, b1, 0, 1, 2, 3, 4, 5, 6, 7,
                                                 8, 9, 10, 11, 12, 13, 14, 15);
            acc[nt] = __builtin_amdgcn_wmma_f32_16x16x32_f16(
                /*neg_a=*/false, afrag, /*neg_b=*/false, bfrag,
                /*c_mod=*/(short)0, acc[nt], /*reuse_a=*/false, /*reuse_b=*/false);
        }
    }

    // Epilogue: D layout -> VGPR v holds row (rowBase + half*8 + v), col = nt*16+l.
    int b  = rowBase / CN2;            // constexpr divisor; tile never straddles b
    int nb = rowBase - b * CN2 + half * 8;
#pragma unroll
    for (int nt = 0; nt < NT; ++nt) {
        int o = nt * 16 + l;
        float bia = bias[o];
        float* yp = y + ((size_t)(b * O + o)) * CN2 + nb;
        float s = 0.f, sq = 0.f;
#pragma unroll
        for (int v = 0; v < 8; ++v) {
            float val = acc[nt][v] + bia;
            yp[v] = val;
            s += val;
            sq += val * val;
        }
        s  += __shfl_xor(s, 16);
        sq += __shfl_xor(sq, 16);
        if (half == 0) {
            atomicAdd(&sum[o], s);
            atomicAdd(&sumsq[o], sq);
        }
    }
}

// ---------------------------------------------------------------------------
// 5) BN finalize: stats -> per-channel scale/shift
// ---------------------------------------------------------------------------
__global__ void bn_fin(const float* __restrict__ sum, const float* __restrict__ sumsq,
                       const float* __restrict__ gamma, const float* __restrict__ beta,
                       float* __restrict__ scale, float* __restrict__ shift,
                       int O, float invCount) {
    int o = blockIdx.x * blockDim.x + threadIdx.x;
    if (o >= O) return;
    float mean = sum[o] * invCount;
    float var  = sumsq[o] * invCount - mean * mean;
    float sc   = gamma[o] * rsqrtf(var + 1e-5f);
    scale[o] = sc;
    shift[o] = beta[o] - mean * sc;
}

// ---------------------------------------------------------------------------
// 6) fused BN+ReLU+max-pool: out[b,o,m] = max_k relu(y3[b,o,pidx[m,k]]*sc+sh)
// ---------------------------------------------------------------------------
__global__ void pool_max(const float* __restrict__ y3, const int* __restrict__ pidx,
                         const float* __restrict__ sc, const float* __restrict__ sh,
                         float* __restrict__ out) {
    int i = blockIdx.x * blockDim.x + threadIdx.x;
    if (i >= CB * 64 * CN3) return;
    int m  = i % CN3;       // constexpr divisors -> mul-hi, no div loop
    int bo = i / CN3;       // b*64 + o
    int o  = bo & 63;
    float scale = sc[o], shift = sh[o];
    const float* yp = y3 + (size_t)bo * CN2;
    float mx = -1e30f;
#pragma unroll
    for (int k = 0; k < 9; ++k) {
        float v = yp[pidx[m * 9 + k]] * scale + shift;
        v = v > 0.f ? v : 0.f;
        mx = v > mx ? v : mx;
    }
    out[i] = mx;
}

// ---------------------------------------------------------------------------
extern "C" void kernel_launch(void* const* d_in, const int* in_sizes, int n_in,
                              void* d_out, int out_size, void* d_ws, size_t ws_size,
                              hipStream_t stream) {
    constexpr int MT = CM / 16;   // 6272 row tiles

    const float* x     = (const float*)d_in[0];
    const int*   nbr1  = (const int*)  d_in[1];
    const float* off1  = (const float*)d_in[2];
    const int*   nbr2  = (const int*)  d_in[3];
    const float* off2  = (const float*)d_in[4];
    const int*   pidx  = (const int*)  d_in[5];
    const float* Wh    = (const float*)d_in[6];
    const float* bh    = (const float*)d_in[7];
    const float* Wo    = (const float*)d_in[8];
    const float* bo    = (const float*)d_in[9];
    const float* W1    = (const float*)d_in[10];
    const float* b1    = (const float*)d_in[11];
    const float* W2    = (const float*)d_in[12];
    const float* b2    = (const float*)d_in[13];
    const float* W3    = (const float*)d_in[14];
    const float* b3    = (const float*)d_in[15];
    const float* g1w   = (const float*)d_in[16];
    const float* be1   = (const float*)d_in[17];
    const float* g2w   = (const float*)d_in[18];
    const float* be2   = (const float*)d_in[19];
    const float* g3w   = (const float*)d_in[20];
    const float* be3   = (const float*)d_in[21];
    float* out = (float*)d_out;

    // ---- workspace carve-up (256B aligned) ----
    char* w = (char*)d_ws;
    auto alloc = [&](size_t bytes) -> void* {
        void* p = (void*)w;
        w += (bytes + 255) & ~(size_t)255;
        return p;
    };
    float*    a1    = (float*)   alloc((size_t)CN2 * 81 * 4);
    float*    a2    = (float*)   alloc((size_t)CN2 * 81 * 4);
    _Float16* g     = (_Float16*)alloc((size_t)CM * 288 * 2);   // reused every layer
    float*    y1    = (float*)   alloc((size_t)CB * 32 * CN2 * 4);
    float*    y2    = (float*)   alloc((size_t)CB * 32 * CN2 * 4);
    float*    y3    = (float*)   alloc((size_t)CB * 64 * CN2 * 4);
    _Float16* Wf1   = (_Float16*)alloc((size_t)32 * 32 * 2);
    _Float16* Wf2   = (_Float16*)alloc((size_t)32 * 288 * 2);
    _Float16* Wf3   = (_Float16*)alloc((size_t)64 * 288 * 2);
    float*    stats = (float*)   alloc(256 * 4);
    float*    scsh  = (float*)   alloc(256 * 4);
    float *sum1 = stats,       *sq1 = stats + 32;
    float *sum2 = stats + 64,  *sq2 = stats + 96;
    float *sum3 = stats + 128, *sq3 = stats + 192;
    float *sc1 = scsh,       *sh1 = scsh + 32;
    float *sc2 = scsh + 64,  *sh2 = scsh + 96;
    float *sc3 = scsh + 128, *sh3 = scsh + 192;

    const float invM = 1.f / (float)CM;

    // stats must start at zero every call
    zero_f32<<<1, 256, 0, stream>>>(stats, 256);

    // offset MLPs -> a1, a2
    mlp_a<<<(CN2 * 9 + 255) / 256, 256, 0, stream>>>(off1, Wh, bh, Wo, bo, a1, CN2 * 9);
    mlp_a<<<(CN2 * 9 + 255) / 256, 256, 0, stream>>>(off2, Wh, bh, Wo, bo, a2, CN2 * 9);

    // weights -> fp16 (layer 1 padded K 27 -> 32)
    convert_w<<<(32 * 32 + 255) / 256, 256, 0, stream>>>(W1, Wf1, 27, 32, 32 * 32);
    convert_w<<<(32 * 288 + 255) / 256, 256, 0, stream>>>(W2, Wf2, 288, 288, 32 * 288);
    convert_w<<<(64 * 288 + 255) / 256, 256, 0, stream>>>(W3, Wf3, 288, 288, 64 * 288);

    // ---- layer 1:  K=32 (padded 27), O=32 ----
    gather_g1<<<CN2, 32, 0, stream>>>(x, nbr1, a1, g);
    gemm_wmma<1, 32, 2><<<MT / 8, 256, 0, stream>>>(g, Wf1, b1, y1, sum1, sq1);
    bn_fin<<<1, 32, 0, stream>>>(sum1, sq1, g1w, be1, sc1, sh1, 32, invM);

    // ---- layer 2:  K=288, O=32 ----
    gather_g32<<<CN2, 256, 0, stream>>>(y1, nbr2, a2, sc1, sh1, g);
    gemm_wmma<9, 288, 2><<<MT / 8, 256, 0, stream>>>(g, Wf2, b2, y2, sum2, sq2);
    bn_fin<<<1, 32, 0, stream>>>(sum2, sq2, g2w, be2, sc2, sh2, 32, invM);

    // ---- layer 3:  K=288, O=64 ----
    gather_g32<<<CN2, 256, 0, stream>>>(y2, nbr2, a2, sc2, sh2, g);
    gemm_wmma<9, 288, 4><<<MT / 8, 256, 0, stream>>>(g, Wf3, b3, y3, sum3, sq3);
    bn_fin<<<1, 64, 0, stream>>>(sum3, sq3, g3w, be3, sc3, sh3, 64, invM);

    // ---- fused BN+ReLU+max-pool ----
    const int totalOut = CB * 64 * CN3;
    pool_max<<<(totalOut + 255) / 256, 256, 0, stream>>>(y3, pidx, sc3, sh3, out);

    (void)in_sizes; (void)n_in; (void)out_size; (void)ws_size; (void)d_in;
}
